// Hybrid_Attention_Delta_1348619731170
// MI455X (gfx1250) — compile-verified, matrix operand order checked by reading, not run on hardware
//
#include <hip/hip_runtime.h>
#include <hip/hip_bf16.h>

// ---------------------------------------------------------------------------
// MI455X (gfx1250, wave32) implementation of the hybrid-attention reference.
// All matmuls run on V_WMMA_F32_16X16X32_F16 (f16 in, f32 accumulate).
// Fragment loads are vectorized: each lane's 16-half WMMA fragment is two
// contiguous 16B runs (ISA 7.12.2), loaded as 2x ds_load_b128/global_load_b128.
// B-side operands are staged TRANSPOSED into LDS so that their fragment reads
// are row-contiguous too.  Pure (conversion-free) global->LDS staging uses
// CDNA5 GLOBAL_LOAD_ASYNC_TO_LDS_B128 tracked by ASYNCcnt.
// ---------------------------------------------------------------------------

typedef __attribute__((ext_vector_type(16))) _Float16 v16h;
typedef __attribute__((ext_vector_type(8)))  _Float16 h8;
typedef __attribute__((ext_vector_type(8)))  float    v8f;

constexpr int Bb     = 8;
constexpr int Nn     = 1024;
constexpr int HDd    = 64;
constexpr int INNERi = 512;

union V16U { v16h v; h8 h2[2]; };

// Async DMA of 16 bytes global -> LDS (ASYNCcnt).  ldsOff is the byte offset
// of the destination within the workgroup's LDS allocation (low 32 bits of
// the generic pointer, per ISA 10.2 aperture mapping).
__device__ __forceinline__ void async_copy_b128(const void* gptr, void* ldsptr) {
  const unsigned ldsOff = (unsigned)(unsigned long long)ldsptr;
  asm volatile("global_load_async_to_lds_b128 %0, %1, off"
               :: "v"(ldsOff), "v"(gptr) : "memory");
}

__device__ __forceinline__ void wait_async0() {
#if __has_builtin(__builtin_amdgcn_s_wait_asynccnt)
  __builtin_amdgcn_s_wait_asynccnt(0);
#else
  asm volatile("s_wait_asynccnt 0x0" ::: "memory");
#endif
}

// Fragment loader: lane holds "line" (lane&15) of a row-major tile; the 16
// halves are K = khi..khi+7 and 16+khi..16+khi+7 with khi = 8*(lane>>4),
// i.e. two aligned 16B vector loads. Serves A-fragments (line = row m),
// K^T-fragments (line = col n) and transposed-staged B-fragments.
__device__ __forceinline__ v16h load_frag(const _Float16* base, int rstride, int lane) {
  const _Float16* p = base + (lane & 15) * rstride + ((lane >> 4) << 3);
  V16U u;
  u.h2[0] = *(const h8*)p;
  u.h2[1] = *(const h8*)(p + 16);
  return u.v;
}

__device__ __forceinline__ v8f wmma_f16(v16h a, v16h b, v8f c) {
  // (neg_a, A, neg_b, B, c_mod, C, reuse_a, reuse_b)
  return __builtin_amdgcn_wmma_f32_16x16x32_f16(false, a, false, b, (short)0, c, false, false);
}

// ---------------------------------------------------------------------------
// GEMM 1: C(f16)[M,Nc] = A(f32)[M,K] @ W(f32)[K,Nc].  BM=BN=128, BK=32.
// 8 waves in a 4x2 grid; each wave owns 32x64 = 2x4 WMMA tiles.
// W tile staged transposed: Bst[col][k].  (f32->f16 conversion forces the
// VGPR staging path here.)
// ---------------------------------------------------------------------------
__global__ __launch_bounds__(256)
void gemm_qkv_kernel(const float* __restrict__ A, const float* __restrict__ W,
                     _Float16* __restrict__ C, int M, int K, int Ncols) {
  constexpr int AS_S = 40;    // 128 rows x 32 halfs (+pad), 16B-multiple stride
  constexpr int BT_S = 40;    // 128 cols x 32 halfs (+pad), transposed W tile
  __shared__ alignas(16) _Float16 As [128 * AS_S];
  __shared__ alignas(16) _Float16 Bst[128 * BT_S];

  const int tid  = threadIdx.x;
  const int lane = tid & 31;
  const int wave = tid >> 5;
  const int row0 = blockIdx.y * 128;
  const int col0 = blockIdx.x * 128;
  const int wm   = wave & 3;   // row group (32 rows)
  const int wn   = wave >> 2;  // col group (64 cols)

  v8f acc[2][4];
#pragma unroll
  for (int i = 0; i < 2; ++i)
#pragma unroll
    for (int nt = 0; nt < 4; ++nt) acc[i][nt] = {};

  for (int k0 = 0; k0 < K; k0 += 32) {
    __syncthreads();
    // Stage A (f32 -> f16), row-major: 128x32 = 1024 float4 chunks.
    for (int i = tid; i < 1024; i += 256) {
      const int r = i >> 3, c4 = (i & 7) << 2;
      const float4 v = *(const float4*)(A + (size_t)(row0 + r) * K + k0 + c4);
      _Float16* d = As + r * AS_S + c4;
      d[0] = (_Float16)v.x; d[1] = (_Float16)v.y;
      d[2] = (_Float16)v.z; d[3] = (_Float16)v.w;
    }
    // Stage W (f32 -> f16) TRANSPOSED: Bst[col][k].
    for (int i = tid; i < 1024; i += 256) {
      const int r = i >> 5, c4 = (i & 31) << 2;  // k-row r, col base c4
      const float4 v = *(const float4*)(W + (size_t)(k0 + r) * Ncols + col0 + c4);
      Bst[(c4 + 0) * BT_S + r] = (_Float16)v.x;
      Bst[(c4 + 1) * BT_S + r] = (_Float16)v.y;
      Bst[(c4 + 2) * BT_S + r] = (_Float16)v.z;
      Bst[(c4 + 3) * BT_S + r] = (_Float16)v.w;
    }
    if (k0 + 32 < K) {  // global_prefetch_b8 of next tiles
      __builtin_prefetch((const void*)(A + (size_t)(row0 + (tid >> 1)) * K + k0 + 32), 0, 1);
      __builtin_prefetch((const void*)(W + (size_t)(k0 + 32 + (tid >> 3)) * Ncols + col0), 0, 1);
    }
    __syncthreads();

    v16h af[2], bf[4];
#pragma unroll
    for (int i = 0; i < 2; ++i)
      af[i] = load_frag(As + (wm * 32 + i * 16) * AS_S, AS_S, lane);
#pragma unroll
    for (int nt = 0; nt < 4; ++nt)
      bf[nt] = load_frag(Bst + (wn * 64 + nt * 16) * BT_S, BT_S, lane);
#pragma unroll
    for (int i = 0; i < 2; ++i)
#pragma unroll
      for (int nt = 0; nt < 4; ++nt)
        acc[i][nt] = wmma_f16(af[i], bf[nt], acc[i][nt]);
  }

  const int hi = lane >> 4, cn = lane & 15;
#pragma unroll
  for (int i = 0; i < 2; ++i)
#pragma unroll
    for (int nt = 0; nt < 4; ++nt)
#pragma unroll
      for (int r = 0; r < 8; ++r) {
        const int row = row0 + wm * 32 + i * 16 + r + 8 * hi;
        const int col = col0 + wn * 64 + nt * 16 + cn;
        C[(size_t)row * Ncols + col] = (_Float16)acc[i][nt][r];
      }
}

// ---------------------------------------------------------------------------
// GEMM 2: out(f32)[M,Nc] = A(f16)[M,K] @ W(f32)[K,Nc] + bias.
// The f16 A tile needs no conversion -> staged with async-to-LDS DMA.
// ---------------------------------------------------------------------------
__global__ __launch_bounds__(256)
void gemm_out_kernel(const _Float16* __restrict__ A, const float* __restrict__ W,
                     const float* __restrict__ bias, float* __restrict__ C,
                     int M, int K, int Ncols) {
  constexpr int AS_S = 40;
  constexpr int BT_S = 40;
  __shared__ alignas(16) _Float16 As [128 * AS_S];
  __shared__ alignas(16) _Float16 Bst[128 * BT_S];

  const int tid  = threadIdx.x;
  const int lane = tid & 31;
  const int wave = tid >> 5;
  const int row0 = blockIdx.y * 128;
  const int col0 = blockIdx.x * 128;
  const int wm   = wave & 3;
  const int wn   = wave >> 2;

  v8f acc[2][4];
#pragma unroll
  for (int i = 0; i < 2; ++i)
#pragma unroll
    for (int nt = 0; nt < 4; ++nt) acc[i][nt] = {};

  for (int k0 = 0; k0 < K; k0 += 32) {
    __syncthreads();
    // Stage A (f16) via async DMA: 128x32 halfs = 512 b128 transfers.
    for (int i = tid; i < 512; i += 256) {
      const int r = i >> 2, c8 = (i & 3) << 3;
      async_copy_b128(A + (size_t)(row0 + r) * K + k0 + c8,
                      As + r * AS_S + c8);
    }
    // Stage W transposed (f32 -> f16 conversion: VGPR path).
    for (int i = tid; i < 1024; i += 256) {
      const int r = i >> 5, c4 = (i & 31) << 2;
      const float4 v = *(const float4*)(W + (size_t)(k0 + r) * Ncols + col0 + c4);
      Bst[(c4 + 0) * BT_S + r] = (_Float16)v.x;
      Bst[(c4 + 1) * BT_S + r] = (_Float16)v.y;
      Bst[(c4 + 2) * BT_S + r] = (_Float16)v.z;
      Bst[(c4 + 3) * BT_S + r] = (_Float16)v.w;
    }
    wait_async0();           // ASYNCcnt == 0: A tile landed in LDS
    __syncthreads();

    v16h af[2], bf[4];
#pragma unroll
    for (int i = 0; i < 2; ++i)
      af[i] = load_frag(As + (wm * 32 + i * 16) * AS_S, AS_S, lane);
#pragma unroll
    for (int nt = 0; nt < 4; ++nt)
      bf[nt] = load_frag(Bst + (wn * 64 + nt * 16) * BT_S, BT_S, lane);
#pragma unroll
    for (int i = 0; i < 2; ++i)
#pragma unroll
      for (int nt = 0; nt < 4; ++nt)
        acc[i][nt] = wmma_f16(af[i], bf[nt], acc[i][nt]);
  }

  const int hi = lane >> 4, cn = lane & 15;
#pragma unroll
  for (int i = 0; i < 2; ++i)
#pragma unroll
    for (int nt = 0; nt < 4; ++nt)
#pragma unroll
      for (int r = 0; r < 8; ++r) {
        const int row = row0 + wm * 32 + i * 16 + r + 8 * hi;
        const int col = col0 + wn * 64 + nt * 16 + cn;
        C[(size_t)row * Ncols + col] = acc[i][nt][r] + bias[col];
      }
}

// ---------------------------------------------------------------------------
// conv_gate: MS = s1+s2 (f32); depthwise conv (k=5, pad 2) along features with
// per-token weights; MSg = sigmoid(conv) * MS -> f16, merged b n (h d) layout.
// One block per (b, n) row.
// ---------------------------------------------------------------------------
__global__ __launch_bounds__(128)
void conv_gate_kernel(const _Float16* __restrict__ qkv1,
                      const _Float16* __restrict__ qkv2,
                      const float* __restrict__ cw, const float* __restrict__ cb,
                      _Float16* __restrict__ msg) {
  const int bn = blockIdx.x;          // b*N + n
  const int n  = bn & (Nn - 1);
  __shared__ float ms[INNERi + 4];

  const _Float16* s1 = qkv1 + (size_t)bn * 2048 + 1536;
  const _Float16* s2 = qkv2 + (size_t)bn * 2048 + 1536;
  for (int f = threadIdx.x; f < INNERi; f += blockDim.x)
    ms[f + 2] = (float)s1[f] + (float)s2[f];
  if (threadIdx.x < 2) { ms[threadIdx.x] = 0.0f; ms[INNERi + 2 + threadIdx.x] = 0.0f; }
  __syncthreads();

  const float w0 = cw[n * 5 + 0], w1 = cw[n * 5 + 1], w2 = cw[n * 5 + 2],
              w3 = cw[n * 5 + 3], w4 = cw[n * 5 + 4];
  const float bias = cb[n];
  for (int f = threadIdx.x; f < INNERi; f += blockDim.x) {
    const float a = w0 * ms[f] + w1 * ms[f + 1] + w2 * ms[f + 2] +
                    w3 * ms[f + 3] + w4 * ms[f + 4] + bias;
    const float g = ms[f + 2] / (1.0f + __expf(-a));
    msg[(size_t)bn * INNERi + f] = (_Float16)g;
  }
}

// ---------------------------------------------------------------------------
// Flash-attention pass: O = softmax(scale * Q K^T) @ V  (+ V residual).
// All tensors live in [B, N, stride] f16 buffers; the head slice is selected
// by (section offset + h*64).  128 query rows per block, 8 waves x 16 rows,
// KV streamed in 64-row chunks through LDS.
//   Ks  : [j][d] row-major, staged via async-to-LDS DMA (no conversion)
//   Vst : [d][j] transposed (VGPR path; transpose forces it)
//   Ps  : per-wave P scratch for C->A relayout
// ---------------------------------------------------------------------------
__global__ __launch_bounds__(256)
void attn_kernel(const _Float16* __restrict__ Q, int qstride, int qsec,
                 const _Float16* __restrict__ Kv, int kstride, int ksec,
                 const _Float16* __restrict__ V, int vstride, int vsec,
                 _Float16* __restrict__ O, int ostride,
                 int addV, float scale) {
  constexpr int LS = 72;  // padded row stride in halfs (144B, 16B-multiple)
  __shared__ alignas(16) _Float16 Ks [64 * LS];
  __shared__ alignas(16) _Float16 Vst[64 * LS];
  __shared__ alignas(16) _Float16 Ps [8 * 16 * LS];

  const int tid  = threadIdx.x;
  const int lane = tid & 31;
  const int wave = tid >> 5;
  const int bh   = blockIdx.y;
  const int b    = bh >> 3;
  const int h    = bh & 7;
  const int row0 = blockIdx.x * 128;
  const size_t bN = (size_t)b * Nn;
  const int qoff = qsec + h * HDd;
  const int koff = ksec + h * HDd;
  const int voff = vsec + h * HDd;
  const int ooff = h * HDd;

  // Q fragments for this wave's 16 rows: two global_load_b128 per fragment.
  v16h qa0, qa1;
  {
    const _Float16* qp =
        Q + (bN + row0 + wave * 16 + (lane & 15)) * (size_t)qstride + qoff +
        ((lane >> 4) << 3);
    V16U u0, u1;
    u0.h2[0] = *(const h8*)(qp);
    u0.h2[1] = *(const h8*)(qp + 16);
    u1.h2[0] = *(const h8*)(qp + 32);
    u1.h2[1] = *(const h8*)(qp + 48);
    qa0 = u0.v;
    qa1 = u1.v;
  }

  float m_old[8], l[8];
#pragma unroll
  for (int r = 0; r < 8; ++r) { m_old[r] = -1e30f; l[r] = 0.0f; }
  v8f o[4];
#pragma unroll
  for (int dt = 0; dt < 4; ++dt) o[dt] = {};

  _Float16* ps = Ps + wave * 16 * LS;
  const int cn = lane & 15;
  const int hi = lane >> 4;

  for (int j0 = 0; j0 < Nn; j0 += 64) {
    __syncthreads();
    // Stage K chunk row-major via async DMA: 64x64 halfs = 512 b128 transfers.
    for (int i = tid; i < 512; i += 256) {
      const int r = i >> 3, c8 = (i & 7) << 3;
      async_copy_b128(Kv + (bN + j0 + r) * (size_t)kstride + koff + c8,
                      Ks + r * LS + c8);
    }
    // Stage V chunk TRANSPOSED: Vst[d][j] (VGPR path).
    for (int i = tid; i < 512; i += 256) {
      const int r = i >> 3, c8 = (i & 7) << 3;  // j row r, d base c8
      const h8 v = *(const h8*)(V + (bN + j0 + r) * (size_t)vstride + voff + c8);
#pragma unroll
      for (int e = 0; e < 8; ++e) Vst[(c8 + e) * LS + r] = v[e];
    }
    if (j0 + 64 < Nn)
      __builtin_prefetch(
          (const void*)(Kv + (bN + j0 + 64 + (tid >> 2)) * (size_t)kstride + koff), 0, 1);
    wait_async0();           // K chunk landed in LDS
    __syncthreads();

    // S = scale * Q K^T  (16x64 per wave, 4 WMMA column tiles, K-depth 64).
    v8f sc[4];
#pragma unroll
    for (int nt = 0; nt < 4; ++nt) {
      v8f z = {};
      z = wmma_f16(qa0, load_frag(Ks + (nt * 16) * LS, LS, lane), z);
      z = wmma_f16(qa1, load_frag(Ks + (nt * 16) * LS + 32, LS, lane), z);
#pragma unroll
      for (int r = 0; r < 8; ++r) z[r] *= scale;
      sc[nt] = z;
    }

    // Online-softmax update.  Stats live per half-wave (rows 0-7 / 8-15).
    float m_new[8], alpha[8];
#pragma unroll
    for (int r = 0; r < 8; ++r) {
      float mx = fmaxf(fmaxf(sc[0][r], sc[1][r]), fmaxf(sc[2][r], sc[3][r]));
#pragma unroll
      for (int off = 1; off < 16; off <<= 1) mx = fmaxf(mx, __shfl_xor(mx, off, 32));
      m_new[r] = fmaxf(m_old[r], mx);
      alpha[r] = __expf(m_old[r] - m_new[r]);
      m_old[r] = m_new[r];
    }
    float rs[8];
#pragma unroll
    for (int r = 0; r < 8; ++r) rs[r] = 0.0f;
#pragma unroll
    for (int nt = 0; nt < 4; ++nt)
#pragma unroll
      for (int r = 0; r < 8; ++r) {
        const float p = __expf(sc[nt][r] - m_new[r]);
        rs[r] += p;
        ps[(r + 8 * hi) * LS + nt * 16 + cn] = (_Float16)p;  // C->A relayout via LDS
      }
#pragma unroll
    for (int r = 0; r < 8; ++r) {
      float s = rs[r];
#pragma unroll
      for (int off = 1; off < 16; off <<= 1) s += __shfl_xor(s, off, 32);
      l[r] = l[r] * alpha[r] + s;
    }
#pragma unroll
    for (int dt = 0; dt < 4; ++dt)
#pragma unroll
      for (int r = 0; r < 8; ++r) o[dt][r] *= alpha[r];

    // O += P @ V_chunk.  A = P from per-wave scratch (row-contiguous);
    // B-fragments from transposed Vst (row-contiguous): all 2x ds_load_b128.
    const v16h pa0 = load_frag(ps, LS, lane);
    const v16h pa1 = load_frag(ps + 32, LS, lane);
#pragma unroll
    for (int dt = 0; dt < 4; ++dt) {
      o[dt] = wmma_f16(pa0, load_frag(Vst + (dt * 16) * LS, LS, lane), o[dt]);
      o[dt] = wmma_f16(pa1, load_frag(Vst + (dt * 16) * LS + 32, LS, lane), o[dt]);
    }
  }

  // Normalize, optional +V residual, store f16.
#pragma unroll
  for (int dt = 0; dt < 4; ++dt)
#pragma unroll
    for (int r = 0; r < 8; ++r) {
      const int grow = row0 + wave * 16 + r + 8 * hi;
      float val = o[dt][r] / l[r];
      if (addV)
        val += (float)V[(bN + grow) * (size_t)vstride + voff + dt * 16 + cn];
      O[(bN + grow) * (size_t)ostride + ooff + dt * 16 + cn] = (_Float16)val;
    }
}

// ---------------------------------------------------------------------------
// Host launcher.
// ---------------------------------------------------------------------------
extern "C" void kernel_launch(void* const* d_in, const int* in_sizes, int n_in,
                              void* d_out, int out_size, void* d_ws, size_t ws_size,
                              hipStream_t stream) {
  (void)in_sizes; (void)n_in; (void)out_size; (void)ws_size;
  const float* x1     = (const float*)d_in[0];
  const float* x2     = (const float*)d_in[1];
  const float* Wqkv1  = (const float*)d_in[2];
  const float* Wqkv2  = (const float*)d_in[3];
  const float* Wout1  = (const float*)d_in[4];
  const float* bout1  = (const float*)d_in[5];
  const float* Wout2  = (const float*)d_in[6];
  const float* bout2  = (const float*)d_in[7];
  const float* conv_w = (const float*)d_in[8];
  const float* conv_b = (const float*)d_in[9];

  const size_t MROWS = (size_t)Bb * Nn;  // 8192
  // Workspace layout (f16). t1 aliases msg (dead after attn pass 2);
  // t2 aliases u1 (dead after attn pass 3). Total 88 MiB.
  _Float16* qkv1h = (_Float16*)d_ws;
  _Float16* qkv2h = qkv1h + MROWS * 2048;
  _Float16* msg   = qkv2h + MROWS * 2048;
  _Float16* u1    = msg + MROWS * 512;
  _Float16* u2    = u1 + MROWS * 512;
  _Float16* t1    = msg;  // alias
  _Float16* t2    = u1;   // alias

  float* out1 = (float*)d_out;
  float* out2 = out1 + MROWS * 512;

  const dim3 blk(256);
  const float scale = 0.125f;  // HD^-0.5

  // QKV projections: [8192,512] @ [512,2048] -> f16
  gemm_qkv_kernel<<<dim3(16, 64), blk, 0, stream>>>(x1, Wqkv1, qkv1h, 8192, 512, 2048);
  gemm_qkv_kernel<<<dim3(16, 64), blk, 0, stream>>>(x2, Wqkv2, qkv2h, 8192, 512, 2048);

  // Gated conv branch -> MSg (merged b n (h d), f16)
  conv_gate_kernel<<<dim3(Bb * Nn), dim3(128), 0, stream>>>(qkv1h, qkv2h, conv_w, conv_b, msg);

  // u1 = softmax(MS MS^T)·v1 + v1 ; u2 likewise (sections: q=0,k=512,v=1024,s=1536)
  attn_kernel<<<dim3(8, 64), blk, 0, stream>>>(msg, 512, 0, msg, 512, 0,
                                               qkv1h, 2048, 1024, u1, 512, 1, scale);
  attn_kernel<<<dim3(8, 64), blk, 0, stream>>>(msg, 512, 0, msg, 512, 0,
                                               qkv2h, 2048, 1024, u2, 512, 1, scale);
  // t1 = softmax(q1 k1^T)·u1 ; t2 = softmax(q2 k2^T)·u2
  attn_kernel<<<dim3(8, 64), blk, 0, stream>>>(qkv1h, 2048, 0, qkv1h, 2048, 512,
                                               u1, 512, 0, t1, 512, 0, scale);
  attn_kernel<<<dim3(8, 64), blk, 0, stream>>>(qkv2h, 2048, 0, qkv2h, 2048, 512,
                                               u2, 512, 0, t2, 512, 0, scale);

  // Output projections with bias -> f32 d_out
  gemm_out_kernel<<<dim3(4, 64), blk, 0, stream>>>(t1, Wout1, bout1, out1, 8192, 512, 512);
  gemm_out_kernel<<<dim3(4, 64), blk, 0, stream>>>(t2, Wout2, bout2, out2, 8192, 512, 512);
}
